// LSCA_45028437131676
// MI455X (gfx1250) — matrix operand order repaired and checked by reading, not compiled, not run on hardware
//
#include <hip/hip_runtime.h>
#include <hip/hip_bf16.h>
#include <math.h>

typedef __attribute__((ext_vector_type(16))) _Float16 v16h;
typedef __attribute__((ext_vector_type(8)))  float    v8f;

#define Bn 4
#define Cc 128
#define Hh 128
#define Wd 128
#define NN (Hh*Wd)
#define NHEADS 8
#define CH 16
#define HID 340
#define H2 680
#define KPO 352   // padded K for pout gemm

// ---------------- weight convert + zero-pad to f16 [Mpad x Kpad] ----------------
__global__ void k_convert_pad(const float* __restrict__ w, _Float16* __restrict__ o,
                              int M, int K, int Mpad, int Kpad) {
  int idx = blockIdx.x * 256 + threadIdx.x;
  if (idx >= Mpad * Kpad) return;
  int m = idx / Kpad, k = idx % Kpad;
  float v = (m < M && k < K) ? w[m * K + k] : 0.f;
  o[idx] = (_Float16)v;
}

// ---------------- channel LayerNorm over C (NCHW), one thread per pixel ----------------
__global__ void k_layernorm(const float* __restrict__ in, const float* __restrict__ g,
                            const float* __restrict__ b, float* __restrict__ out) {
  int idx = blockIdx.x * 256 + threadIdx.x;   // Bn*NN threads
  int bb = idx / NN, p = idx % NN;
  size_t base = (size_t)bb * Cc * NN + p;
  float s = 0.f, s2 = 0.f;
  for (int c = 0; c < Cc; ++c) { float v = in[base + (size_t)c * NN]; s += v; s2 += v * v; }
  float mu = s * (1.f / Cc);
  float var = s2 * (1.f / Cc) - mu * mu;
  float rs = rsqrtf(var + 1e-6f);
  for (int c = 0; c < Cc; ++c) {
    float v = in[base + (size_t)c * NN];
    out[base + (size_t)c * NN] = g[c] * (v - mu) * rs + b[c];
  }
}

// ---------------- 1x1 conv as WMMA GEMM: Out[b,m,n] = sum_k Wp[m,k]*In[b,k,n] ----------------
// In has exactly KPAD channels per batch (activations pre-padded where needed), so the
// K-loop is guard-free and fully unrolled: loads clause up, one wait per WMMA step.
template<int KPAD>
__global__ void k_gemm_wmma(const _Float16* __restrict__ Wp, const float* __restrict__ In,
                            float* __restrict__ Out, const float* __restrict__ bias,
                            const float* __restrict__ add, int M, int relu) {
  int lane = threadIdx.x & 31;
  int wave = threadIdx.x >> 5;
  int ntile = blockIdx.x * 4 + wave;
  int mtile = blockIdx.y;
  int bb = blockIdx.z;
  int half = lane >> 4, lm = lane & 15;
  int ncol = ntile * 16 + lm;
  size_t inBase  = (size_t)bb * KPAD * NN + ncol;
  size_t outBase = (size_t)bb * M * NN + ncol;
  const _Float16* Wrow = Wp + (size_t)(mtile * 16 + lm) * KPAD;
  v8f acc = {};
#pragma unroll
  for (int kb = 0; kb < KPAD; kb += 32) {
    if (kb + 32 < KPAD)
      __builtin_prefetch(&In[inBase + (size_t)(kb + 32) * NN], 0, 3);
    v16h a, bm;
#pragma unroll
    for (int e = 0; e < 16; ++e) {                      // A 16x32 f16, contiguous k = 16*half + e
      int ka = kb + e + 16 * half;
      a[e] = Wrow[ka];
    }
#pragma unroll
    for (int e = 0; e < 16; ++e) {                      // B 32x16 f16 layout, guard-free
      int kc = kb + e + 16 * half;
      bm[e] = (_Float16)In[inBase + (size_t)kc * NN];
    }
    acc = __builtin_amdgcn_wmma_f32_16x16x32_f16(false, a, false, bm, (short)0, acc, false, false);
  }
#pragma unroll
  for (int r = 0; r < 8; ++r) {                          // C/D: row = r + 8*half, col = lm
    int m = mtile * 16 + r + 8 * half;
    if (m < M) {
      float v = acc[r];
      if (bias) v += bias[m];
      if (relu) v = fmaxf(v, 0.f);
      size_t o = outBase + (size_t)m * NN;
      if (add) v += add[o];
      Out[o] = v;
    }
  }
}

// ---------------- depthwise 3x3, pad=1; mode 1: out = tanh(conv) + in_center ----------------
__global__ void k_dwconv3(const float* __restrict__ in, const float* __restrict__ w9,
                          float* __restrict__ out, int Cn, size_t inBS, size_t outBS, int mode) {
  size_t idx = (size_t)blockIdx.x * 256 + threadIdx.x;
  size_t tot = (size_t)Bn * Cn * NN;
  if (idx >= tot) return;
  int p = (int)(idx % NN); size_t t = idx / NN;
  int c = (int)(t % Cn); int bb = (int)(t / Cn);
  int x = p % Wd, y = p / Wd;
  const float* ip = in + (size_t)bb * inBS + (size_t)c * NN;
  const float* wc = w9 + c * 9;
  float s;
  if (x >= 1 && x < Wd - 1 && y >= 1 && y < Hh - 1) {
    // interior fast path: unguarded 9-point stencil (loads clause up)
    const float* r0 = ip + (y - 1) * Wd + (x - 1);
    const float* r1 = r0 + Wd;
    const float* r2 = r1 + Wd;
    s = wc[0] * r0[0] + wc[1] * r0[1] + wc[2] * r0[2]
      + wc[3] * r1[0] + wc[4] * r1[1] + wc[5] * r1[2]
      + wc[6] * r2[0] + wc[7] * r2[1] + wc[8] * r2[2];
  } else {
    s = 0.f;
#pragma unroll
    for (int ky = -1; ky <= 1; ++ky)
#pragma unroll
      for (int kx = -1; kx <= 1; ++kx) {
        int yy = y + ky, xx = x + kx;
        if (yy >= 0 && yy < Hh && xx >= 0 && xx < Wd)
          s += wc[(ky + 1) * 3 + (kx + 1)] * ip[yy * Wd + xx];
      }
  }
  float r = (mode == 1) ? (tanhf(s) + ip[p]) : s;
  out[(size_t)bb * outBS + (size_t)c * NN + p] = r;
}

// ---------------- L2 row normalization over N=16384, one block per row ----------------
__global__ void k_rownorm(float* __restrict__ d, int chPerB, size_t bStride) {
  int row = blockIdx.x;                 // Bn*chPerB rows
  int bb = row / chPerB, c = row % chPerB;
  float* p = d + (size_t)bb * bStride + (size_t)c * NN;
  __shared__ float red[256];
  float s = 0.f;
  for (int i = threadIdx.x; i < NN; i += 256) { float v = p[i]; s += v * v; }
  red[threadIdx.x] = s; __syncthreads();
  for (int st = 128; st > 0; st >>= 1) {
    if (threadIdx.x < st) red[threadIdx.x] += red[threadIdx.x + st];
    __syncthreads();
  }
  float sc = 1.f / fmaxf(sqrtf(red[0]), 1e-12f);
  for (int i = threadIdx.x; i < NN; i += 256) p[i] *= sc;
}

// ---------------- channel-attention scores: S[16,16] = sum_n q[c,n]k[d,n], WMMA K-loop ----------------
__global__ void k_attn_score(const float* __restrict__ q, const float* __restrict__ kk,
                             const float* __restrict__ temp, float* __restrict__ S) {
  int h = blockIdx.x, bb = blockIdx.y;
  int lane = threadIdx.x & 31;
  int half = lane >> 4, lm = lane & 15;
  const float* qb  = q  + ((size_t)bb * Cc     + h * CH) * NN;
  const float* kbp = kk + ((size_t)bb * 2 * Cc + h * CH) * NN;
  v8f acc = {};
  for (int n0 = 0; n0 < NN; n0 += 32) {
    v16h a, bm;
#pragma unroll
    for (int e = 0; e < 16; ++e) {
      int n = n0 + e + 16 * half;
      a[e] = (_Float16)qb[(size_t)lm * NN + n];        // A row = q channel lm
    }
#pragma unroll
    for (int e = 0; e < 16; ++e) {
      int n = n0 + e + 16 * half;
      bm[e] = (_Float16)kbp[(size_t)lm * NN + n];      // B col = k channel lm
    }
    acc = __builtin_amdgcn_wmma_f32_16x16x32_f16(false, a, false, bm, (short)0, acc, false, false);
  }
  float tt = temp[h];
  float* Sb = S + ((size_t)bb * NHEADS + h) * 256;
#pragma unroll
  for (int r = 0; r < 8; ++r)
    Sb[(r + 8 * half) * 16 + lm] = acc[r] * tt;
}

// ---------------- gate stage 2: per-pixel sigmoid(g2 . relu(g1(x))) -> block partial sums ----------------
__global__ void k_gate2(const float* __restrict__ g1o, const float* __restrict__ g2w,
                        const float* __restrict__ g2b, float* __restrict__ red) {
  int idx = blockIdx.x * 256 + threadIdx.x;  // Bn*NN
  int bb = idx / NN, p = idx % NN;
  const float* gp = g1o + (size_t)bb * 64 * NN + p;
  float s = 0.f;
  for (int c = 0; c < 64; ++c) s += gp[(size_t)c * NN] * g2w[c];
  s += g2b[0];
  float sig = 1.f / (1.f + expf(-s));
  __shared__ float sh[256];
  sh[threadIdx.x] = sig; __syncthreads();
  for (int st = 128; st > 0; st >>= 1) {
    if (threadIdx.x < st) sh[threadIdx.x] += sh[threadIdx.x + st];
    __syncthreads();
  }
  if (threadIdx.x == 0) red[blockIdx.x] = sh[0];
}

__global__ void k_dk(const float* __restrict__ red, int* __restrict__ dk) {
  __shared__ float sh[256];
  sh[threadIdx.x] = red[threadIdx.x]; __syncthreads();
  for (int st = 128; st > 0; st >>= 1) {
    if (threadIdx.x < st) sh[threadIdx.x] += sh[threadIdx.x + st];
    __syncthreads();
  }
  if (threadIdx.x == 0) {
    float mean = sh[0] / (float)(Bn * NN);
    int v = (int)(CH * mean);
    if (v < 1) v = 1;
    if (v > CH) v = CH;
    dk[0] = v;
  }
}

// ---------------- top-k mask (rank < dk, stable-tie by index) + row softmax, 16x16 per head ----------------
__global__ void k_masksoftmax(float* __restrict__ S, const int* __restrict__ dkp) {
  int h = blockIdx.x, bb = blockIdx.y;
  float* Sb = S + ((size_t)bb * NHEADS + h) * 256;
  __shared__ float s[256]; __shared__ float m[256];
  int t = threadIdx.x; int c = t >> 4, d = t & 15;
  s[t] = Sb[t]; __syncthreads();
  int dk = dkp[0];
  float mine = s[c * 16 + d];
  int rank = 0;
#pragma unroll
  for (int j = 0; j < 16; ++j) {
    float o = s[c * 16 + j];
    rank += (o > mine) || (o == mine && j < d);
  }
  m[t] = (rank < dk) ? mine : -INFINITY; __syncthreads();
  float mx = -INFINITY;
#pragma unroll
  for (int j = 0; j < 16; ++j) mx = fmaxf(mx, m[c * 16 + j]);
  float num = (m[t] == -INFINITY) ? 0.f : expf(m[t] - mx);
  __syncthreads(); s[t] = num; __syncthreads();
  float sum = 0.f;
#pragma unroll
  for (int j = 0; j < 16; ++j) sum += s[c * 16 + j];
  Sb[t] = num / sum;
}

// ---------------- out[c,n] = sum_d P[c,d] * v[d,n] (P in LDS) ----------------
__global__ void k_attn_apply(const float* __restrict__ P, const float* __restrict__ v,
                             float* __restrict__ out) {
  int h = blockIdx.y, bb = blockIdx.z;
  int n = blockIdx.x * 256 + threadIdx.x;
  __shared__ float p[256];
  p[threadIdx.x] = P[((size_t)bb * NHEADS + h) * 256 + threadIdx.x];
  __syncthreads();
  const float* vb = v + ((size_t)bb * 2 * Cc + Cc + h * CH) * NN + n;
  float acc[16];
#pragma unroll
  for (int c = 0; c < 16; ++c) acc[c] = 0.f;
  for (int d = 0; d < 16; ++d) {
    float vv = vb[(size_t)d * NN];
#pragma unroll
    for (int c = 0; c < 16; ++c) acc[c] += p[c * 16 + d] * vv;
  }
  float* ob = out + ((size_t)bb * Cc + h * CH) * NN + n;
#pragma unroll
  for (int c = 0; c < 16; ++c) ob[(size_t)c * NN] = acc[c];
}

// ---- prod[b,c,n] = x1t[b,c,n]*x2t[b,c,n] into compact zero-padded [Bn, KPO, NN] ----
__global__ void k_mulpad(const float* __restrict__ a, float* __restrict__ o) {
  size_t idx = (size_t)blockIdx.x * 256 + threadIdx.x;   // Bn*KPO*NN
  size_t per = (size_t)KPO * NN;
  size_t tot = (size_t)Bn * per;
  if (idx >= tot) return;
  size_t bb = idx / per, r = idx % per;
  const float* base = a + bb * (size_t)H2 * NN;
  float v = 0.f;
  if (r < (size_t)HID * NN) v = base[r] * base[(size_t)HID * NN + r];
  o[bb * per + r] = v;
}

extern "C" void kernel_launch(void* const* d_in, const int* in_sizes, int n_in,
                              void* d_out, int out_size, void* d_ws, size_t ws_size,
                              hipStream_t stream) {
  (void)in_sizes; (void)n_in; (void)out_size; (void)ws_size;
  const float* x     = (const float*)d_in[0];
  const float* y     = (const float*)d_in[1];
  const float* ln_w  = (const float*)d_in[2];
  const float* ln_b  = (const float*)d_in[3];
  const float* temp  = (const float*)d_in[4];
  const float* q_w   = (const float*)d_in[5];
  const float* qdw_w = (const float*)d_in[6];
  const float* kv_w  = (const float*)d_in[7];
  const float* kvdw_w= (const float*)d_in[8];
  const float* po_w  = (const float*)d_in[9];
  const float* g1_w  = (const float*)d_in[10];
  const float* g1_b  = (const float*)d_in[11];
  const float* g2_w  = (const float*)d_in[12];
  const float* g2_b  = (const float*)d_in[13];
  const float* pin_w = (const float*)d_in[14];
  const float* dw_w  = (const float*)d_in[15];
  const float* dw1_w = (const float*)d_in[16];
  const float* dw2_w = (const float*)d_in[17];
  const float* pout_w= (const float*)d_in[18];
  float* out = (float*)d_out;

  const size_t szC   = (size_t)Bn * Cc * NN;
  const size_t szKV  = (size_t)Bn * 2 * Cc * NN;
  const size_t szBig = (size_t)Bn * H2 * NN;
  const size_t szG   = (size_t)Bn * 64 * NN;

  float* W = (float*)d_ws;
  size_t off = 0;
  auto alloc = [&](size_t n) { float* p = W + off; off += n; return p; };
  float* xn   = alloc(szC);
  float* yn   = alloc(szC);
  float* R1   = alloc(2 * szC + 2 * szKV);   // phase1: q|qd|kv|kvd ; phase2: t (szBig fits)
  float* td   = alloc(szBig);                // dw output; later: compact padded prod
  float* gbuf = alloc(szG);
  float* aout = alloc(szC);
  float* xatt = alloc(szC);
  float* Sbuf = alloc((size_t)Bn * NHEADS * 256);
  float* red  = alloc(512);
  int*   dk   = (int*)alloc(16);
  _Float16* hw = (_Float16*)alloc(131072);

  _Float16* Wq    = hw;
  _Float16* Wkv   = Wq   + 16384;   // 128x128
  _Float16* Wpo   = Wkv  + 32768;   // 256x128
  _Float16* Wg1   = Wpo  + 16384;   // 128x128
  _Float16* Wpin  = Wg1  + 8192;    // 64x128
  _Float16* Wpout = Wpin + 88064;   // 688x128 ; pout 128x352

  float* qb  = R1;
  float* qd  = R1 + szC;
  float* kv  = R1 + 2 * szC;
  float* kvd = R1 + 2 * szC + szKV;
  float* tb  = R1;                  // phase-2 alias (q/kv dead by then)

  auto cvt = [&](const float* src, _Float16* dst, int M, int K, int Mp, int Kp) {
    int tot = Mp * Kp;
    k_convert_pad<<<(tot + 255) / 256, 256, 0, stream>>>(src, dst, M, K, Mp, Kp);
  };
  cvt(q_w,    Wq,    128, 128, 128, 128);
  cvt(kv_w,   Wkv,   256, 128, 256, 128);
  cvt(po_w,   Wpo,   128, 128, 128, 128);
  cvt(g1_w,   Wg1,    64, 128,  64, 128);
  cvt(pin_w,  Wpin,  680, 128, 688, 128);
  cvt(pout_w, Wpout, 128, 340, 128, KPO);

  k_layernorm<<<Bn * NN / 256, 256, 0, stream>>>(x, ln_w, ln_b, xn);
  k_layernorm<<<Bn * NN / 256, 256, 0, stream>>>(y, ln_w, ln_b, yn);

  dim3 gblk(128);
  auto grid = [&](int Mp) { return dim3(NN / 64, Mp / 16, Bn); };

  k_gemm_wmma<128><<<grid(128), gblk, 0, stream>>>(Wq,  xn, qb,   nullptr, nullptr, 128, 0);
  k_gemm_wmma<128><<<grid(256), gblk, 0, stream>>>(Wkv, yn, kv,   nullptr, nullptr, 256, 0);
  k_gemm_wmma<128><<<grid(64),  gblk, 0, stream>>>(Wg1, xn, gbuf, g1_b,    nullptr,  64, 1);

  auto dwc = [&](const float* in, const float* w, float* o, int Cn,
                 size_t bsIn, size_t bsOut, int mode) {
    size_t tot = (size_t)Bn * Cn * NN;
    k_dwconv3<<<(unsigned)((tot + 255) / 256), 256, 0, stream>>>(in, w, o, Cn, bsIn, bsOut, mode);
  };
  dwc(qb, qdw_w,  qd,  128, (size_t)128 * NN, (size_t)128 * NN, 0);
  dwc(kv, kvdw_w, kvd, 256, (size_t)256 * NN, (size_t)256 * NN, 0);

  k_gate2<<<Bn * NN / 256, 256, 0, stream>>>(gbuf, g2_w, g2_b, red);
  k_dk<<<1, 256, 0, stream>>>(red, dk);

  k_rownorm<<<Bn * Cc, 256, 0, stream>>>(qd,  Cc, (size_t)Cc * NN);       // q rows
  k_rownorm<<<Bn * Cc, 256, 0, stream>>>(kvd, Cc, (size_t)2 * Cc * NN);   // k rows only

  k_attn_score<<<dim3(NHEADS, Bn), 32, 0, stream>>>(qd, kvd, temp, Sbuf);
  k_masksoftmax<<<dim3(NHEADS, Bn), 256, 0, stream>>>(Sbuf, dk);
  k_attn_apply<<<dim3(NN / 256, NHEADS, Bn), 256, 0, stream>>>(Sbuf, kvd, aout);

  k_gemm_wmma<128><<<grid(128), gblk, 0, stream>>>(Wpo, aout, xatt, nullptr, x, 128, 0);

  // ---- IEL ----
  k_layernorm<<<Bn * NN / 256, 256, 0, stream>>>(xatt, ln_w, ln_b, xn);   // z -> xn (reuse)
  k_gemm_wmma<128><<<grid(688), gblk, 0, stream>>>(Wpin, xn, tb, nullptr, nullptr, 680, 0);
  dwc(tb, dw_w, td, 680, (size_t)H2 * NN, (size_t)H2 * NN, 0);            // td = dw(t)
  dwc(td, dw1_w, tb, 340, (size_t)H2 * NN, (size_t)H2 * NN, 1);           // x1t = tanh(dw1)+x1
  dwc(td + (size_t)HID * NN, dw2_w, tb + (size_t)HID * NN, 340,
      (size_t)H2 * NN, (size_t)H2 * NN, 1);                               // x2t
  {
    size_t tot = (size_t)Bn * KPO * NN;
    k_mulpad<<<(unsigned)((tot + 255) / 256), 256, 0, stream>>>(tb, td);  // padded prod
  }
  k_gemm_wmma<KPO><<<grid(128), gblk, 0, stream>>>(Wpout, td, out, nullptr, xatt, 128, 0);
}
